// MemoryModule_29875792511776
// MI455X (gfx1250) — matrix operand order repaired and checked by reading, not compiled, not run on hardware
//
#include <hip/hip_runtime.h>
#include <hip/hip_bf16.h>

typedef __attribute__((ext_vector_type(16))) __bf16 v16bf;
typedef __attribute__((ext_vector_type(8)))  __bf16 v8bf;
typedef __attribute__((ext_vector_type(8)))  float  v8f;
typedef __attribute__((ext_vector_type(4)))  float  v4f;
typedef __attribute__((ext_vector_type(4)))  int    v4i;

#define B_TOTAL 65536
#define DK 512
#define DV 512
#define DH 64
#define WAVES 4
#define TPB (WAVES * 32)
#define NB (DV / 16)
#define PSTR (DK + 8)   // padded LDS row stride (bf16 elems)
#define HSTR (DH + 8)

// workspace layout (bf16 elements)
#define WS_W1_OFF   0
#define WS_W2_OFF   (DH * DK)                 // 32768
#define WS_PREC_OFF (DH * DK + DV * DH)       // 65536
#define WS_ELEMS    (WS_PREC_OFF + DV * DK)   // 327680
#define WS_BYTES    (WS_ELEMS * 2)

#if defined(__has_builtin)
#if __has_builtin(__builtin_amdgcn_global_load_async_to_lds_b128)
#define HAVE_ASYNC_LDS 1
#endif
#endif
#ifndef HAVE_ASYNC_LDS
#define HAVE_ASYNC_LDS 0
#endif

#if defined(__has_builtin)
#if __has_builtin(__builtin_amdgcn_s_wait_asynccnt)
#define WAIT_ASYNC(n) __builtin_amdgcn_s_wait_asynccnt(n)
#endif
#endif
#ifndef WAIT_ASYNC
#define WAIT_ASYNC(n) asm volatile("s_wait_asynccnt %0" ::"i"(n) : "memory")
#endif

static __device__ __forceinline__ v8f wmma_bf16(v16bf a, v16bf b, v8f c) {
  return __builtin_amdgcn_wmma_f32_16x16x32_bf16(false, a, false, b, (short)0, c,
                                                 false, false);
}

// ---- A-operand tile (16x32 MxK): lane m holds contiguous K c0+g*16..+15. ----
static __device__ __forceinline__ v16bf atile_from_f32_nt(const float* __restrict__ base,
                                                          int stride, int lane, int c0) {
  const int g = lane >> 4, r = lane & 15;
  const float* p = base + (size_t)r * stride + c0 + g * 16;
  v4f a = __builtin_nontemporal_load((const v4f*)p);
  v4f b = __builtin_nontemporal_load(((const v4f*)p) + 1);
  v4f c = __builtin_nontemporal_load((const v4f*)(p + 8));
  v4f d = __builtin_nontemporal_load(((const v4f*)(p + 8)) + 1);
  v16bf t;
#pragma unroll
  for (int i = 0; i < 4; ++i) {
    t[i] = (__bf16)a[i]; t[4 + i] = (__bf16)b[i];
    t[8 + i] = (__bf16)c[i]; t[12 + i] = (__bf16)d[i];
  }
  return t;
}

static __device__ __forceinline__ v16bf atile_from_lds(const __bf16* base, int stride,
                                                       int lane, int c0) {
  const int g = lane >> 4, r = lane & 15;
  const __bf16* p = base + r * stride + c0 + g * 16;
  v8bf lo = *(const v8bf*)p;
  v8bf hi = *(const v8bf*)(p + 8);
  return __builtin_shufflevector(lo, hi, 0, 1, 2, 3, 4, 5, 6, 7,
                                 8, 9, 10, 11, 12, 13, 14, 15);
}

// ---- B-operand tile (32x16 KxN): lane n holds contiguous K c0+g*16..+15. ----
static __device__ __forceinline__ v16bf btile_from_f32(const float* __restrict__ base,
                                                       int stride, int lane, int c0) {
  const float* p = base + (size_t)(lane & 15) * stride + c0 + (lane >> 4) * 16;
  v16bf t;
#pragma unroll
  for (int q = 0; q < 4; ++q) {
    v4f v = ((const v4f*)p)[q];
#pragma unroll
    for (int i = 0; i < 4; ++i) t[q * 4 + i] = (__bf16)v[i];
  }
  return t;
}

static __device__ __forceinline__ v16bf btile_from_bf16g(const __bf16* __restrict__ base,
                                                         int stride, int lane, int c0) {
  const __bf16* p = base + (size_t)(lane & 15) * stride + c0 + (lane >> 4) * 16;
  v8bf lo = *(const v8bf*)p;
  v8bf hi = *(const v8bf*)(p + 8);
  return __builtin_shufflevector(lo, hi, 0, 1, 2, 3, 4, 5, 6, 7,
                                 8, 9, 10, 11, 12, 13, 14, 15);
}

static __device__ __forceinline__ v16bf btile_from_lds(const __bf16* base, int stride,
                                                       int lane, int c0) {
  const __bf16* p = base + (lane & 15) * stride + c0 + (lane >> 4) * 16;
  v8bf lo = *(const v8bf*)p;
  v8bf hi = *(const v8bf*)(p + 8);
  return __builtin_shufflevector(lo, hi, 0, 1, 2, 3, 4, 5, 6, 7,
                                 8, 9, 10, 11, 12, 13, 14, 15);
}

#if HAVE_ASYNC_LDS
static __device__ __forceinline__ void cp_async16(const __bf16* g, __bf16* l) {
  __builtin_amdgcn_global_load_async_to_lds_b128(
      (__attribute__((address_space(1))) v4i*)g,
      (__attribute__((address_space(3))) v4i*)l, 0, 0);
}
#endif

__global__ void convert_weights(const float* __restrict__ W1,
                                const float* __restrict__ W2,
                                const float* __restrict__ pW,
                                __bf16* __restrict__ ws) {
  int i = (int)blockIdx.x * 256 + (int)threadIdx.x;
  float v;
  if (i < WS_W2_OFF) v = W1[i];
  else if (i < WS_PREC_OFF) v = W2[i - WS_W2_OFF];
  else v = pW[i - WS_PREC_OFF];
  ws[i] = (__bf16)v;
}

template <bool PRE>
__global__ __launch_bounds__(TPB) void memmod_kernel(
    const float* __restrict__ key, const float* __restrict__ value,
    const float* __restrict__ W1, const float* __restrict__ b1,
    const float* __restrict__ W2, const float* __restrict__ b2,
    const float* __restrict__ precW, const float* __restrict__ pb,
    float* __restrict__ out, const __bf16* __restrict__ ws) {
  __shared__ alignas(32) __bf16 sPrec[2][16][PSTR];     // 2 x 16.6 KB (double buffer)
  __shared__ alignas(32) __bf16 sW2[2][16][HSTR];       // 2 x  2.3 KB
  __shared__ alignas(32) __bf16 sH[WAVES][16][HSTR];    //      9.2 KB

  const int tid  = threadIdx.x;
  const int lane = tid & 31;
  const int wv   = tid >> 5;
  const int row0 = ((int)blockIdx.x * WAVES + wv) * 16;
  const int g    = lane >> 4;
  const int nlo  = lane & 15;

  // ---- key block (16 x 512) -> bf16 A-layout tiles, resident in VGPRs ----
  v16bf kA[DK / 32];
#pragma unroll
  for (int kt = 0; kt < DK / 32; ++kt)
    kA[kt] = atile_from_f32_nt(key + (size_t)row0 * DK, DK, lane, kt * 32);

  // ---- phase 1: h = gelu(key @ W1^T + b1)  (16 x 64) ----
#pragma unroll
  for (int nb1 = 0; nb1 < DH / 16; ++nb1) {
    v8f acc = {};
#pragma unroll
    for (int kt = 0; kt < DK / 32; ++kt) {
      v16bf bt;
      if constexpr (PRE)
        bt = btile_from_bf16g(ws + WS_W1_OFF + (size_t)(nb1 * 16) * DK, DK, lane, kt * 32);
      else
        bt = btile_from_f32(W1 + (size_t)(nb1 * 16) * DK, DK, lane, kt * 32);
      acc = wmma_bf16(kA[kt], bt, acc);
    }
    const int n = nb1 * 16 + nlo;
    const float bias = b1[n];
#pragma unroll
    for (int r = 0; r < 8; ++r) {
      float x = acc[r] + bias;
      float hg = 0.5f * x * (1.0f + erff(x * 0.70710678118654752f));
      sH[wv][r + 8 * g][n] = (__bf16)hg;   // C-layout -> row-major via LDS
    }
  }
  v16bf hA0 = atile_from_lds(&sH[wv][0][0], HSTR, lane, 0);
  v16bf hA1 = atile_from_lds(&sH[wv][0][0], HSTR, lane, 32);

  float fsum[8];
#pragma unroll
  for (int r = 0; r < 8; ++r) fsum[r] = 0.0f;

  const size_t outPred = (size_t)B_TOTAL;
  const size_t outPrec = outPred + (size_t)B_TOTAL * DV;
  const size_t outErr  = outPrec + (size_t)B_TOTAL * DV;

  auto compute_block = [&](int nb, const __bf16 (*bp)[PSTR], const __bf16 (*bw)[HSTR]) {
    v8f accP = {};
    accP = wmma_bf16(hA0, btile_from_lds(&bw[0][0], HSTR, lane, 0), accP);
    accP = wmma_bf16(hA1, btile_from_lds(&bw[0][0], HSTR, lane, 32), accP);
    v8f accR = {};
#pragma unroll
    for (int kt = 0; kt < DK / 32; ++kt)
      accR = wmma_bf16(kA[kt], btile_from_lds(&bp[0][0], PSTR, lane, kt * 32), accR);

    const int n = nb * 16 + nlo;
    const float bias2 = b2[n];
    const float biasp = pb[n];
#pragma unroll
    for (int r = 0; r < 8; ++r) {
      const size_t grow = (size_t)(row0 + r + 8 * g);
      float predv = accP[r] + bias2;
      float x = accR[r] + biasp;
      float sp = (x > 20.0f) ? x : log1pf(__expf(x));   // softplus
      float prec = sp + 0.01f;
      float val = __builtin_nontemporal_load(&value[grow * DV + n]);
      float err = fminf(3.0f, fmaxf(-3.0f, val - predv));
      fsum[r] += prec * err * err - __logf(prec);
      __builtin_nontemporal_store(predv, &out[outPred + grow * DV + n]);
      __builtin_nontemporal_store(prec,  &out[outPrec + grow * DV + n]);
      __builtin_nontemporal_store(err,   &out[outErr  + grow * DV + n]);
    }
  };

  if constexpr (PRE) {
    const __bf16* gP = ws + WS_PREC_OFF;
    const __bf16* gW = ws + WS_W2_OFF;
#if HAVE_ASYNC_LDS
    // double-buffered async staging: LDS fill overlapped with WMMA compute
    auto stage_async = [&](int nb, __bf16 (*bp)[PSTR], __bf16 (*bw)[HSTR]) {
#pragma unroll
      for (int it = 0; it < 8; ++it) {              // 8 x b128 per lane: prec block
        int ch = it * TPB + tid;
        int rr = ch >> 6, cc = (ch & 63) * 8;
        cp_async16(gP + (size_t)(nb * 16 + rr) * DK + cc, &bp[rr][cc]);
      }
      {                                             // 1 x b128 per lane: W2 block
        int rr = tid >> 3, cc = (tid & 7) * 8;
        cp_async16(gW + (size_t)(nb * 16 + rr) * DH + cc, &bw[rr][cc]);
      }
    };
    stage_async(0, sPrec[0], sW2[0]);
    for (int nb = 0; nb < NB; ++nb) {
      const int cur = nb & 1;
      if (nb + 1 < NB) {
        stage_async(nb + 1, sPrec[1 - cur], sW2[1 - cur]);
        WAIT_ASYNC(9);     // 9 ops of stage(nb+1) in flight; stage(nb) retired
      } else {
        WAIT_ASYNC(0);
      }
      __syncthreads();
      compute_block(nb, sPrec[cur], sW2[cur]);
      __syncthreads();
    }
#else
    for (int nb = 0; nb < NB; ++nb) {
      __syncthreads();
#pragma unroll
      for (int it = 0; it < 8; ++it) {
        int ch = it * TPB + tid;
        int rr = ch >> 6, cc = (ch & 63) * 8;
        *(v8bf*)&sPrec[0][rr][cc] = *(const v8bf*)(gP + (size_t)(nb * 16 + rr) * DK + cc);
      }
      {
        int rr = tid >> 3, cc = (tid & 7) * 8;
        *(v8bf*)&sW2[0][rr][cc] = *(const v8bf*)(gW + (size_t)(nb * 16 + rr) * DH + cc);
      }
      __syncthreads();
      compute_block(nb, sPrec[0], sW2[0]);
    }
#endif
  } else {
    // fallback: stage f32 weights with on-the-fly conversion
    for (int nb = 0; nb < NB; ++nb) {
      __syncthreads();
      for (int i = tid; i < 16 * DK; i += TPB) {
        int rr = i >> 9, cc = i & (DK - 1);
        sPrec[0][rr][cc] = (__bf16)precW[(size_t)(nb * 16 + rr) * DK + cc];
      }
      for (int i = tid; i < 16 * DH; i += TPB) {
        int rr = i >> 6, cc = i & (DH - 1);
        sW2[0][rr][cc] = (__bf16)W2[(size_t)(nb * 16 + rr) * DH + cc];
      }
      if (nb + 1 < NB)
        __builtin_prefetch(&precW[(size_t)((nb + 1) * 16) * DK + tid * 64], 0, 1);
      __syncthreads();
      compute_block(nb, sPrec[0], sW2[0]);
    }
  }

  // F = row mean: reduce partials across the 16 lanes of each half-wave
#pragma unroll
  for (int r = 0; r < 8; ++r) {
#pragma unroll
    for (int off = 8; off >= 1; off >>= 1)
      fsum[r] += __shfl_xor(fsum[r], off, 32);
  }
  if (nlo == 0) {
#pragma unroll
    for (int r = 0; r < 8; ++r)
      out[row0 + r + 8 * g] = fsum[r] * (1.0f / 512.0f);
  }
}

extern "C" void kernel_launch(void* const* d_in, const int* in_sizes, int n_in,
                              void* d_out, int out_size, void* d_ws, size_t ws_size,
                              hipStream_t stream) {
  const float* key = (const float*)d_in[0];
  const float* val = (const float*)d_in[1];
  const float* W1  = (const float*)d_in[2];
  const float* b1  = (const float*)d_in[3];
  const float* W2  = (const float*)d_in[4];
  const float* b2  = (const float*)d_in[5];
  const float* pW  = (const float*)d_in[6];
  const float* pbv = (const float*)d_in[7];
  float* out = (float*)d_out;
  (void)in_sizes; (void)n_in; (void)out_size;

  dim3 grid(B_TOTAL / (WAVES * 16));
  const bool usePre = (d_ws != nullptr) && (ws_size >= (size_t)WS_BYTES);
  if (usePre) {
    __bf16* ws = (__bf16*)d_ws;
    convert_weights<<<dim3(WS_ELEMS / 256), dim3(256), 0, stream>>>(W1, W2, pW, ws);
    memmod_kernel<true><<<grid, dim3(TPB), 0, stream>>>(key, val, W1, b1, W2, b2, pW,
                                                        pbv, out, ws);
  } else {
    memmod_kernel<false><<<grid, dim3(TPB), 0, stream>>>(key, val, W1, b1, W2, b2, pW,
                                                         pbv, out, nullptr);
  }
}
